// GPT_7679401526029
// MI455X (gfx1250) — matrix operand order repaired and checked by reading, not compiled, not run on hardware
//
#include <hip/hip_runtime.h>
#include <hip/hip_bf16.h>
#include <math.h>

// ---------------- model dims ----------------
#define VOCAB 50304
#define TSEQ  1024
#define NB    2
#define NL    4
#define NH    12
#define CCH   768
#define HDIM  64
#define MTOT  (NB * TSEQ)   // 2048

typedef __attribute__((ext_vector_type(16))) __bf16 v16bf;
typedef __attribute__((ext_vector_type(8)))  float  v8f;

// ---------------- bf16 helpers ----------------
static __device__ __forceinline__ __bf16 f2bf(float f) {
    unsigned u = __builtin_bit_cast(unsigned, f);
    unsigned r = (u + 0x7FFFu + ((u >> 16) & 1u)) >> 16;   // RNE
    unsigned short s = (unsigned short)r;
    return __builtin_bit_cast(__bf16, s);
}
static __device__ __forceinline__ float bf2f(__bf16 b) {
    unsigned short s = __builtin_bit_cast(unsigned short, b);
    unsigned u = ((unsigned)s) << 16;
    return __builtin_bit_cast(float, u);
}

// ---------------- CDNA5 async global->LDS copy (ASYNCcnt path) ----------------
// GLOBAL_LOAD_ASYNC_TO_LDS_B128: VDST = LDS byte address (low 32 bits of generic
// LDS pointer per aperture mapping), VADDR = 64-bit global address, SADDR = off.
static __device__ __forceinline__ void async_copy_b128(void* lds_dst, const void* gsrc) {
    unsigned l = (unsigned)(size_t)lds_dst;
    unsigned long long g = (unsigned long long)(size_t)gsrc;
    asm volatile("global_load_async_to_lds_b128 %0, %1, off" :: "v"(l), "v"(g) : "memory");
}
static __device__ __forceinline__ void async_wait0() {
    asm volatile("s_wait_asynccnt 0" ::: "memory");
}

// ---------------- WMMA wrapper ----------------
static __device__ __forceinline__ v8f wmma_bf16(v16bf a, v16bf b, v8f c) {
    return __builtin_amdgcn_wmma_f32_16x16x32_bf16(false, a, false, b, (short)0, c,
                                                   false, false);
}

// A fragment: 16x32 bf16. lane m = L%16, half = L/16;
// elem e -> K = (e&7) + (e>>3)*16 + half*8  (both 8-runs contiguous -> b128 loads)
static __device__ __forceinline__ v16bf frag_a_lds(const __bf16* base, int stride, int lane) {
    int m = lane & 15, half = lane >> 4;
    const __bf16* p = base + m * stride + half * 8;
    v16bf f;
#pragma unroll
    for (int e = 0; e < 8; ++e) f[e] = p[e];
#pragma unroll
    for (int e = 0; e < 8; ++e) f[8 + e] = p[16 + e];
    return f;
}

// B fragment from a TRANSPOSED LDS tile laid out [N][K] (stride in elems).
// lane n = L%16, half = L/16; elem e -> K = (e&7) + (e>>3)*16 + half*8 (matches A frag).
static __device__ __forceinline__ v16bf frag_b_ldsT(const __bf16* base, int stride, int lane) {
    int n = lane & 15, half = lane >> 4;
    const __bf16* p = base + n * stride + half * 8;
    v16bf f;
#pragma unroll
    for (int e = 0; e < 16; ++e) f[e] = p[(e & 7) + (e >> 3) * 16];
    return f;
}

// ---------------- embedding ----------------
__global__ __launch_bounds__(256) void embed_kernel(const int* __restrict__ idx,
                                                    const float* __restrict__ wte,
                                                    const float* __restrict__ wpe,
                                                    float* __restrict__ x) {
    int row = blockIdx.x;
    int t = row % TSEQ;
    int tok = idx[row];
    for (int c = threadIdx.x; c < CCH; c += 256)
        x[(size_t)row * CCH + c] = wte[(size_t)tok * CCH + c] + wpe[(size_t)t * CCH + c];
}

// ---------------- layernorm -> bf16 ----------------
__global__ __launch_bounds__(256) void ln_kernel(const float* __restrict__ x,
                                                 const float* __restrict__ w,
                                                 __bf16* __restrict__ out) {
    __shared__ float red[256];
    int row = blockIdx.x, tid = threadIdx.x;
    const float* xr = x + (size_t)row * CCH;
    float s = 0.f;
    for (int c = tid; c < CCH; c += 256) s += xr[c];
    red[tid] = s; __syncthreads();
    for (int o = 128; o > 0; o >>= 1) { if (tid < o) red[tid] += red[tid + o]; __syncthreads(); }
    float mu = red[0] / (float)CCH;
    __syncthreads();
    float v = 0.f;
    for (int c = tid; c < CCH; c += 256) { float d = xr[c] - mu; v += d * d; }
    red[tid] = v; __syncthreads();
    for (int o = 128; o > 0; o >>= 1) { if (tid < o) red[tid] += red[tid + o]; __syncthreads(); }
    float rstd = rsqrtf(red[0] / (float)CCH + 1e-5f);
    __syncthreads();
    for (int c = tid; c < CCH; c += 256)
        out[(size_t)row * CCH + c] = f2bf((xr[c] - mu) * rstd * w[c]);
}

// ---------------- tiled WMMA GEMM (128x128 block tile, K-step 32) ----------------
// C[M,N] = A_bf16[M,K] @ W_f32 ; TRANSB=false: W is [K,N]; true: W is [N,K] (wte^T)
// A tile staged via async global->LDS copies; weight tile staged TRANSPOSED [128 n][40 k].
// EPI: 0 -> outF = acc ; 1 -> outF += acc ; 2 -> outB = bf16(gelu(acc)) ; 3 -> outB = bf16(acc)
template <int EPI, bool TRANSB>
__global__ __launch_bounds__(256) void gemm_kernel(const __bf16* __restrict__ A,
                                                   const float* __restrict__ W,
                                                   float* __restrict__ outF,
                                                   __bf16* __restrict__ outB,
                                                   int M, int N, int K) {
    __shared__ __bf16 ldsA[128 * 40];   // 128 m x 32 k (pad 40)
    __shared__ __bf16 ldsB[128 * 40];   // 128 n x 32 k (pad 40), TRANSPOSED
    int tid = threadIdx.x, lane = tid & 31, wave = tid >> 5;
    int m0 = blockIdx.x * 128, n0 = blockIdx.y * 128;

    v8f acc[8];
#pragma unroll
    for (int g = 0; g < 8; ++g)
#pragma unroll
        for (int r = 0; r < 8; ++r) acc[g][r] = 0.f;

    for (int k0 = 0; k0 < K; k0 += 32) {
        { // A tile: 128x32 bf16 via async copies (2x b128 per thread)
            int row = tid >> 1, ch = tid & 1;
            const __bf16* g = A + (size_t)(m0 + row) * K + k0 + ch * 16;
            __bf16* sp = ldsA + row * 40 + ch * 16;
            async_copy_b128(sp, g);
            async_copy_b128(sp + 8, g + 8);
            if (k0 + 32 < K)
                __builtin_prefetch(A + (size_t)(m0 + row) * K + (k0 + 32) + ch * 16, 0, 3);
        }
        if (!TRANSB) { // W[K,N]: thread reads 16 f32 along n, scatters transposed into LDS
            int kr = tid >> 3, cs = (tid & 7) * 16;
            const float* g = W + (size_t)(k0 + kr) * N + n0 + cs;
#pragma unroll
            for (int j = 0; j < 16; ++j) ldsB[(cs + j) * 40 + kr] = f2bf(g[j]);
        } else {       // W[N,K] (wte): reads 16 f32 along k -> contiguous vector store
            int n = tid >> 1, ks = (tid & 1) * 16;
            const float* g = W + (size_t)(n0 + n) * K + k0 + ks;
            __attribute__((aligned(16))) __bf16 tmp[16];
#pragma unroll
            for (int j = 0; j < 16; ++j) tmp[j] = f2bf(g[j]);
            uint4* sp = (uint4*)(ldsB + n * 40 + ks);
            sp[0] = ((const uint4*)tmp)[0];
            sp[1] = ((const uint4*)tmp)[1];
            if (k0 + 32 < K)
                __builtin_prefetch(W + (size_t)(n0 + n) * K + (k0 + 32) + ks, 0, 3);
        }
        async_wait0();      // this wave's async copies are in LDS
        __syncthreads();    // all waves' staging visible
        v16bf af = frag_a_lds(ldsA + wave * 16 * 40, 40, lane);
#pragma unroll
        for (int g = 0; g < 8; ++g) {
            v16bf bf = frag_b_ldsT(ldsB + g * 16 * 40, 40, lane);
            acc[g] = wmma_bf16(af, bf, acc[g]);
        }
        __syncthreads();
    }

    int half = lane >> 4, n = lane & 15;
#pragma unroll
    for (int g = 0; g < 8; ++g)
#pragma unroll
        for (int r = 0; r < 8; ++r) {
            int row = m0 + wave * 16 + r + half * 8;
            int col = n0 + g * 16 + n;
            size_t o = (size_t)row * N + col;
            float v = acc[g][r];
            if (EPI == 0) outF[o] = v;
            else if (EPI == 1) outF[o] += v;
            else if (EPI == 2) {
                float gl = 0.5f * v * (1.0f + erff(v * 0.70710678118f));
                outB[o] = f2bf(gl);
            } else outB[o] = f2bf(v);
        }
}

// ---------------- fused RBF attention ----------------
// qkv bf16 [2048, 3C]; per head: S = exp(scale*max(q2+k2-2qk,0)) masked, y = S @ v
__global__ __launch_bounds__(256) void attn_kernel(const __bf16* __restrict__ qkv,
                                                   __bf16* __restrict__ yout) {
    __shared__ __bf16 Kt[32 * 72];      // [key][hd] : score B-frag reads contiguous in hd
    __shared__ __bf16 VtT[64 * 40];     // [hd][key] TRANSPOSED: S@V B-frag contiguous in key
    __shared__ float  k2s[32];
    __shared__ float  q2part[256];
    __shared__ __bf16 Sst[8 * 16 * 34]; // per-wave 16x32 score staging (pad 34)

    int tid = threadIdx.x, lane = tid & 31, wave = tid >> 5;
    int bh = blockIdx.x, b = bh / NH, h = bh % NH;
    int q0 = blockIdx.y * 128;
    int m = lane & 15, half = lane >> 4;

    // load Q fragments once from global, accumulate q^2 partials
    int qrow = q0 + wave * 16 + m;
    const __bf16* qp = qkv + (size_t)(b * TSEQ + qrow) * (3 * CCH) + h * HDIM;
    v16bf qa0, qa1;
    float part = 0.f;
#pragma unroll
    for (int e = 0; e < 16; ++e) {
        int kk = (e & 7) + ((e >> 3) * 16) + half * 8;
        __bf16 x0 = qp[kk], x1 = qp[kk + 32];
        qa0[e] = x0; qa1[e] = x1;
        float f0 = bf2f(x0), f1 = bf2f(x1);
        part += f0 * f0 + f1 * f1;
    }
    q2part[wave * 32 + lane] = part;   // wave-private; DS in-order within wave
    float q2r[8];
#pragma unroll
    for (int r = 0; r < 8; ++r) {
        int mr = r + half * 8;
        q2r[r] = q2part[wave * 32 + mr] + q2part[wave * 32 + 16 + mr];
    }

    v8f yacc[4];
#pragma unroll
    for (int g = 0; g < 4; ++g)
#pragma unroll
        for (int r = 0; r < 8; ++r) yacc[g][r] = 0.f;

    const float scale = -0.0625f;  // -1/(2*sqrt(64))
    int kend = q0 + 128;
    for (int kc = 0; kc < kend; kc += 32) {
        { // K tile via async copy; V transposed via DS stores; k^2 via 8-lane shuffles
            int kr = tid >> 3, cs = (tid & 7) * 8;
            const __bf16* kg = qkv + (size_t)(b * TSEQ + kc + kr) * (3 * CCH) + CCH + h * HDIM + cs;
            const __bf16* vg = qkv + (size_t)(b * TSEQ + kc + kr) * (3 * CCH) + 2 * CCH + h * HDIM + cs;
            async_copy_b128(Kt + kr * 72 + cs, kg);
#pragma unroll
            for (int j = 0; j < 8; ++j) VtT[(cs + j) * 40 + kr] = vg[j];  // transpose V
            float p2 = 0.f;
#pragma unroll
            for (int j = 0; j < 8; ++j) { float f = bf2f(kg[j]); p2 += f * f; }
            p2 += __shfl_down(p2, 4, 8);
            p2 += __shfl_down(p2, 2, 8);
            p2 += __shfl_down(p2, 1, 8);
            if ((tid & 7) == 0) k2s[kr] = p2;
        }
        async_wait0();
        __syncthreads();

        // scores: two 16-key groups, each 2 WMMAs over HD halves
#pragma unroll
        for (int g2 = 0; g2 < 2; ++g2) {
            v8f s;
#pragma unroll
            for (int r = 0; r < 8; ++r) s[r] = 0.f;
            v16bf kb0, kb1;  // B = K^T: elem e -> hd = (e&7) + (e>>3)*16 + half*8, key col = g2*16 + m
#pragma unroll
            for (int e = 0; e < 16; ++e) {
                int hd = (e & 7) + (e >> 3) * 16 + half * 8;
                kb0[e] = Kt[(g2 * 16 + m) * 72 + hd];
                kb1[e] = Kt[(g2 * 16 + m) * 72 + 32 + hd];
            }
            s = wmma_bf16(qa0, kb0, s);
            s = wmma_bf16(qa1, kb1, s);
#pragma unroll
            for (int r = 0; r < 8; ++r) {
                int qq = q0 + wave * 16 + r + half * 8;
                int kkey = kc + g2 * 16 + m;
                float d = q2r[r] + k2s[g2 * 16 + m] - 2.0f * s[r];
                d = fmaxf(d, 0.f);
                float gv = (kkey <= qq) ? __expf(scale * d) : 0.f;
                Sst[wave * 16 * 34 + (r + half * 8) * 34 + g2 * 16 + m] = f2bf(gv);
            }
        }

        // y += S(16x32) @ V(32x64): 4 WMMAs, B frags contiguous from VtT
        v16bf sa = frag_a_lds(Sst + wave * 16 * 34, 34, lane);
#pragma unroll
        for (int g = 0; g < 4; ++g) {
            v16bf vb = frag_b_ldsT(VtT + g * 16 * 40, 40, lane);
            yacc[g] = wmma_bf16(sa, vb, yacc[g]);
        }
        __syncthreads();
    }

#pragma unroll
    for (int g = 0; g < 4; ++g)
#pragma unroll
        for (int r = 0; r < 8; ++r) {
            int row = q0 + wave * 16 + r + half * 8;
            yout[(size_t)(b * TSEQ + row) * CCH + h * HDIM + g * 16 + m] = f2bf(yacc[g][r]);
        }
}

// ---------------- loss ----------------
__global__ __launch_bounds__(256) void loss_row_kernel(const float* __restrict__ logits,
                                                       const int* __restrict__ targets,
                                                       float* __restrict__ nll) {
    __shared__ float red[256];
    int row = blockIdx.x, tid = threadIdx.x;
    const float* lg = logits + (size_t)row * VOCAB;
    float mx = -3.4e38f;
    for (int v = tid; v < VOCAB; v += 256) mx = fmaxf(mx, lg[v]);
    red[tid] = mx; __syncthreads();
    for (int o = 128; o > 0; o >>= 1) { if (tid < o) red[tid] = fmaxf(red[tid], red[tid + o]); __syncthreads(); }
    mx = red[0]; __syncthreads();
    float se = 0.f;
    for (int v = tid; v < VOCAB; v += 256) se += __expf(lg[v] - mx);
    red[tid] = se; __syncthreads();
    for (int o = 128; o > 0; o >>= 1) { if (tid < o) red[tid] += red[tid + o]; __syncthreads(); }
    if (tid == 0) {
        int t = targets[row];
        float v = 0.f;
        if (t >= 0) v = (mx + logf(red[0])) - lg[t];
        nll[row] = v;
    }
}

__global__ __launch_bounds__(256) void loss_final_kernel(const float* __restrict__ nll,
                                                         const int* __restrict__ targets,
                                                         float* __restrict__ loss_out) {
    __shared__ float red[256];
    __shared__ int   cnt[256];
    int tid = threadIdx.x;
    float s = 0.f; int c = 0;
    for (int r = tid; r < MTOT; r += 256) { s += nll[r]; c += (targets[r] >= 0) ? 1 : 0; }
    red[tid] = s; cnt[tid] = c; __syncthreads();
    for (int o = 128; o > 0; o >>= 1) {
        if (tid < o) { red[tid] += red[tid + o]; cnt[tid] += cnt[tid + o]; }
        __syncthreads();
    }
    if (tid == 0) loss_out[0] = red[0] / fmaxf((float)cnt[0], 1.f);
}

// ---------------- host launch ----------------
extern "C" void kernel_launch(void* const* d_in, const int* in_sizes, int n_in,
                              void* d_out, int out_size, void* d_ws, size_t ws_size,
                              hipStream_t stream) {
    const int*   idx         = (const int*)d_in[0];
    const int*   targets     = (const int*)d_in[1];
    const float* wte         = (const float*)d_in[2];
    const float* wpe         = (const float*)d_in[3];
    const float* ln1_w       = (const float*)d_in[4];
    const float* attn_w      = (const float*)d_in[5];
    const float* attn_proj_w = (const float*)d_in[6];
    const float* ln2_w       = (const float*)d_in[7];
    const float* fc_w        = (const float*)d_in[8];
    const float* fc_proj_w   = (const float*)d_in[9];
    const float* lnf_w       = (const float*)d_in[10];
    (void)in_sizes; (void)n_in; (void)out_size; (void)ws_size;

    char* ws = (char*)d_ws;
    float*  x    = (float*)(ws);                                    // 2048*768 f32
    __bf16* lnb  = (__bf16*)(ws + 6291456);                         // 2048*768 bf16
    __bf16* qkvb = (__bf16*)(ws + 6291456 + 3145728);               // 2048*2304 bf16
    __bf16* yatt = (__bf16*)(ws + 6291456 + 3145728 + 9437184);     // 2048*768 bf16
    __bf16* fca  = (__bf16*)(ws + 6291456 + 3145728 + 9437184 + 3145728);  // 2048*3072 bf16
    float*  nllp = (float*)(ws + 6291456 + 3145728 + 9437184 + 3145728 + 12582912);

    float* logits = (float*)d_out;
    float* lossp  = logits + (size_t)MTOT * VOCAB;

    embed_kernel<<<MTOT, 256, 0, stream>>>(idx, wte, wpe, x);

    for (int l = 0; l < NL; ++l) {
        ln_kernel<<<MTOT, 256, 0, stream>>>(x, ln1_w + (size_t)l * CCH, lnb);
        gemm_kernel<3, false><<<dim3(MTOT / 128, (3 * CCH) / 128), 256, 0, stream>>>(
            lnb, attn_w + (size_t)l * CCH * 3 * CCH, nullptr, qkvb, MTOT, 3 * CCH, CCH);
        attn_kernel<<<dim3(NB * NH, TSEQ / 128), 256, 0, stream>>>(qkvb, yatt);
        gemm_kernel<1, false><<<dim3(MTOT / 128, CCH / 128), 256, 0, stream>>>(
            yatt, attn_proj_w + (size_t)l * CCH * CCH, x, nullptr, MTOT, CCH, CCH);
        ln_kernel<<<MTOT, 256, 0, stream>>>(x, ln2_w + (size_t)l * CCH, lnb);
        gemm_kernel<2, false><<<dim3(MTOT / 128, (4 * CCH) / 128), 256, 0, stream>>>(
            lnb, fc_w + (size_t)l * CCH * 4 * CCH, nullptr, fca, MTOT, 4 * CCH, CCH);
        gemm_kernel<1, false><<<dim3(MTOT / 128, CCH / 128), 256, 0, stream>>>(
            fca, fc_proj_w + (size_t)l * 4 * CCH * CCH, x, nullptr, MTOT, CCH, 4 * CCH);
    }

    ln_kernel<<<MTOT, 256, 0, stream>>>(x, lnf_w, lnb);
    gemm_kernel<0, true><<<dim3(MTOT / 128, VOCAB / 128), 256, 0, stream>>>(
        lnb, wte, logits, nullptr, MTOT, VOCAB, CCH);

    loss_row_kernel<<<MTOT, 256, 0, stream>>>(logits, targets, nllp);
    loss_final_kernel<<<1, 256, 0, stream>>>(nllp, targets, lossp);
}
